// AccumulatingBiLinearKernel_85615878078672
// MI455X (gfx1250) — matrix operand order repaired and checked by reading, not compiled, not run on hardware
//
#include <hip/hip_runtime.h>
#include <math.h>

// ---------------------------------------------------------------------------
// Problem dims (fixed by the reference setup_inputs)
// ---------------------------------------------------------------------------
#define B_DIM 64
#define Q_DIM 512
#define S_DIM 1024
#define D_DIM 1024

typedef __attribute__((ext_vector_type(16))) _Float16 v16h;
typedef __attribute__((ext_vector_type(4)))  _Float16 v4h;
typedef __attribute__((ext_vector_type(8)))  float    v8f;

// GEMM tiling: 128x128 C tile per 256-thread block, K-step 32, double-buffered.
#define TM 128
#define TN 128
#define TK 32
#define LDS_STRIDE 36  // 32 + pad, keeps 4B/8B alignment, spreads banks

union Frag16 {
  v16h h;
  unsigned int u[8];
};

// Split one float4 into f16 hi/lo and store as two 8-byte LDS chunks.
__device__ __forceinline__ void split_store(_Float16* hiP, _Float16* loP, float4 f) {
  v4h h, l;
  h.x = (_Float16)f.x; h.y = (_Float16)f.y; h.z = (_Float16)f.z; h.w = (_Float16)f.w;
  l.x = (_Float16)(f.x - (float)h.x);
  l.y = (_Float16)(f.y - (float)h.y);
  l.z = (_Float16)(f.z - (float)h.z);
  l.w = (_Float16)(f.w - (float)h.w);
  *(v4h*)hiP = h;
  *(v4h*)loP = l;
}

// ---------------------------------------------------------------------------
// Phase 1: scores[b][q][s] = (query[q,b,:] . key[b,s,:]) / 32, via f16x3 WMMA
// (hi*hi + hi*lo + lo*hi compensated split, ~f32 accuracy).
// Stored directly into d_out in the final [B,Q,S] attention layout.
// Double-buffered LDS: next K-tile's HBM loads overlap current tile's WMMAs.
// ---------------------------------------------------------------------------
__global__ __launch_bounds__(256)
void scores_wmma_kernel(const float* __restrict__ key,     // [B,S,D]
                        const float* __restrict__ query,   // [Q,B,D]
                        float* __restrict__ scores) {      // [B,Q,S]
  __shared__ _Float16 sA_hi[2][TM][LDS_STRIDE];
  __shared__ _Float16 sA_lo[2][TM][LDS_STRIDE];
  __shared__ _Float16 sB_hi[2][TN][LDS_STRIDE];
  __shared__ _Float16 sB_lo[2][TN][LDS_STRIDE];

  const int b   = blockIdx.z;
  const int q0  = blockIdx.y * TM;
  const int s0  = blockIdx.x * TN;
  const int tid = threadIdx.x;
  const int lane = tid & 31;
  const int wave = tid >> 5;
  const int wm = (wave >> 2) * 64;  // wave M origin in tile (2 waves along M)
  const int wn = (wave & 3) * 32;   // wave N origin in tile (4 waves along N)

  // Per-thread cooperative-load geometry: 128 rows x 8 float4 per tile,
  // 4 (row,col) slots per thread for each of A and B.
  int ldR[4], ldC[4];
#pragma unroll
  for (int i = 0; i < 4; ++i) {
    const int idx = i * 256 + tid;     // 0..1023
    ldR[i] = idx >> 3;
    ldC[i] = (idx & 7) << 2;
  }

  v8f c[8];
  {
    v8f z = {};
#pragma unroll
    for (int i = 0; i < 8; ++i) c[i] = z;
  }

  const int ml    = lane & 15;
  const int koffA = (lane & 16) ? 8 : 0;    // A: lanes 16-31 hold K+8 (ISA 7.12.2)
  const int kbB   = (lane & 16) ? 8 : 0;    // B: lanes 16-31 hold K+8

  // --- prologue: fetch tile kk=0 and stage into buffer 0
  float4 va[4], vb[4];
#pragma unroll
  for (int i = 0; i < 4; ++i) {
    va[i] = *(const float4*)(query + ((size_t)(q0 + ldR[i]) * B_DIM + b) * D_DIM + ldC[i]);
    vb[i] = *(const float4*)(key   + ((size_t)b * S_DIM + (s0 + ldR[i])) * D_DIM + ldC[i]);
  }
#pragma unroll
  for (int i = 0; i < 4; ++i) {
    split_store(&sA_hi[0][ldR[i]][ldC[i]], &sA_lo[0][ldR[i]][ldC[i]], va[i]);
    split_store(&sB_hi[0][ldR[i]][ldC[i]], &sB_lo[0][ldR[i]][ldC[i]], vb[i]);
  }
  __syncthreads();

  int buf = 0;
  for (int kk = 0; kk < D_DIM; kk += TK) {
    const bool more = (kk + TK) < D_DIM;

    // --- issue next tile's global loads early (overlaps WMMAs below)
    if (more) {
      const int kn = kk + TK;
#pragma unroll
      for (int i = 0; i < 4; ++i) {
        va[i] = *(const float4*)(query + ((size_t)(q0 + ldR[i]) * B_DIM + b) * D_DIM + kn + ldC[i]);
        vb[i] = *(const float4*)(key   + ((size_t)b * S_DIM + (s0 + ldR[i])) * D_DIM + kn + ldC[i]);
      }
    }

    // --- load fragments from current buffer (per-lane ISA 7.12.2 layouts)
    Frag16 aH[4], aL[4], bH[2], bL[2];
#pragma unroll
    for (int mi = 0; mi < 4; ++mi) {
      const int m = wm + mi * 16 + ml;
#pragma unroll
      for (int v = 0; v < 8; ++v) {
        const int k = ((v & 4) << 2) + koffA + ((v & 3) << 1);
        aH[mi].u[v] = *(const unsigned int*)&sA_hi[buf][m][k];
        aL[mi].u[v] = *(const unsigned int*)&sA_lo[buf][m][k];
      }
    }
#pragma unroll
    for (int ni = 0; ni < 2; ++ni) {
      const int n = wn + ni * 16 + ml;
#pragma unroll
      for (int v = 0; v < 8; ++v) {
        const int k = ((v & 4) << 2) + kbB + ((v & 3) << 1);
        bH[ni].u[v] = *(const unsigned int*)&sB_hi[buf][n][k];
        bL[ni].u[v] = *(const unsigned int*)&sB_lo[buf][n][k];
      }
    }

    // --- f16x3 compensated matmul
#pragma unroll
    for (int mi = 0; mi < 4; ++mi) {
#pragma unroll
      for (int ni = 0; ni < 2; ++ni) {
        const int ci = mi * 2 + ni;
        c[ci] = __builtin_amdgcn_wmma_f32_16x16x32_f16(
            false, aH[mi].h, false, bH[ni].h, (short)0, c[ci], false, false);
        c[ci] = __builtin_amdgcn_wmma_f32_16x16x32_f16(
            false, aH[mi].h, false, bL[ni].h, (short)0, c[ci], false, false);
        c[ci] = __builtin_amdgcn_wmma_f32_16x16x32_f16(
            false, aL[mi].h, false, bH[ni].h, (short)0, c[ci], false, false);
      }
    }

    // --- stage next tile into alternate buffer; one barrier per K-step.
    // All reads of `buf` above precede this barrier; stores into `buf`
    // happen only in the *next* iteration, after this barrier, so this is safe.
    if (more) {
      const int nb = buf ^ 1;
#pragma unroll
      for (int i = 0; i < 4; ++i) {
        split_store(&sA_hi[nb][ldR[i]][ldC[i]], &sA_lo[nb][ldR[i]][ldC[i]], va[i]);
        split_store(&sB_hi[nb][ldR[i]][ldC[i]], &sB_lo[nb][ldR[i]][ldC[i]], vb[i]);
      }
      __syncthreads();
      buf = nb;
    }
  }

  // --- write back: C layout VGPR r -> M = r + 8*(lane>=16), N = lane&15
  const float scale = 0.03125f;  // 1/sqrt(1024)
  const int mhi = (lane & 16) ? 8 : 0;
#pragma unroll
  for (int mi = 0; mi < 4; ++mi) {
#pragma unroll
    for (int ni = 0; ni < 2; ++ni) {
      const v8f cv = c[mi * 2 + ni];
      const int s = s0 + wn + ni * 16 + ml;
#pragma unroll
      for (int r = 0; r < 8; ++r) {
        const int q = q0 + wm + mi * 16 + mhi + r;
        scores[((size_t)b * Q_DIM + q) * S_DIM + s] = cv[r] * scale;
      }
    }
  }
}

// ---------------------------------------------------------------------------
// Phase 2: serial softmax scan over q. One wave32 per batch b; each lane owns
// 32 columns of acc in registers. Rows read+overwritten in place in d_out.
// Next row is prefetched to hide memory latency behind exp + reductions.
// ---------------------------------------------------------------------------
__global__ __launch_bounds__(32)
void scan_softmax_kernel(const unsigned char* __restrict__ key_mask,  // [B,S] bool
                         float* __restrict__ out) {                   // [B,Q,S] + [B,S]
  const int b    = blockIdx.x;
  const int lane = threadIdx.x;

  float acc[32], mneg[32];
#pragma unroll
  for (int j = 0; j < 32; ++j) {
    acc[j]  = 0.0f;
    mneg[j] = key_mask[b * S_DIM + j * 32 + lane] ? -INFINITY : 0.0f;
  }

  float* attn_b = out + (size_t)b * Q_DIM * S_DIM;

  float xq[32];
#pragma unroll
  for (int j = 0; j < 32; ++j) xq[j] = attn_b[j * 32 + lane];  // row 0

  for (int q = 0; q < Q_DIM; ++q) {
    float* row = attn_b + (size_t)q * S_DIM;

    // prefetch next row (independent of acc) to overlap with compute
    float xn[32];
    if (q + 1 < Q_DIM) {
      const float* nrow = row + S_DIM;
#pragma unroll
      for (int j = 0; j < 32; ++j) xn[j] = nrow[j * 32 + lane];
    }

    // mod = score - acc  (+ -inf where masked), stabilized softmax over S
    float lmax = -INFINITY;
    float x[32];
#pragma unroll
    for (int j = 0; j < 32; ++j) {
      x[j] = xq[j] - acc[j] + mneg[j];
      lmax = fmaxf(lmax, x[j]);
    }
#pragma unroll
    for (int o = 16; o > 0; o >>= 1) lmax = fmaxf(lmax, __shfl_xor(lmax, o, 32));

    float lsum = 0.0f;
#pragma unroll
    for (int j = 0; j < 32; ++j) {
      x[j] = __expf(x[j] - lmax);
      lsum += x[j];
    }
#pragma unroll
    for (int o = 16; o > 0; o >>= 1) lsum += __shfl_xor(lsum, o, 32);

    const float inv = 1.0f / lsum;
#pragma unroll
    for (int j = 0; j < 32; ++j) {
      const float a = x[j] * inv;
      row[j * 32 + lane] = a;   // overwrite score row with attention row
      acc[j] += a;
    }

#pragma unroll
    for (int j = 0; j < 32; ++j) xq[j] = xn[j];
  }

  // state accumulator output: [B,1,S] appended after attention
  float* accOut = out + (size_t)B_DIM * Q_DIM * S_DIM + (size_t)b * S_DIM;
#pragma unroll
  for (int j = 0; j < 32; ++j) accOut[j * 32 + lane] = acc[j];
}

// ---------------------------------------------------------------------------
extern "C" void kernel_launch(void* const* d_in, const int* in_sizes, int n_in,
                              void* d_out, int out_size, void* d_ws, size_t ws_size,
                              hipStream_t stream) {
  (void)in_sizes; (void)n_in; (void)d_ws; (void)ws_size; (void)out_size;
  const float*         key      = (const float*)d_in[0];          // [B,S,D] f32
  const float*         query    = (const float*)d_in[1];          // [Q,B,D] f32
  const unsigned char* key_mask = (const unsigned char*)d_in[2];  // [B,S] bool
  float* out = (float*)d_out;

  dim3 grid(S_DIM / TN, Q_DIM / TM, B_DIM);  // (8, 4, 64)
  scores_wmma_kernel<<<grid, dim3(256), 0, stream>>>(key, query, out);
  scan_softmax_kernel<<<dim3(B_DIM), dim3(32), 0, stream>>>(key_mask, out);
}
